// CRF_91233695302211
// MI455X (gfx1250) — matrix operand order repaired and checked by reading, not compiled, not run on hardware
//
#include <hip/hip_runtime.h>
#include <hip/hip_bf16.h>

#define SEQL 512
#define BATCH 1024
#define NT 48   // tags

typedef _Float16 half16 __attribute__((ext_vector_type(16)));
typedef _Float16 half8  __attribute__((ext_vector_type(8)));
typedef float    float8 __attribute__((ext_vector_type(8)));
typedef float    float4v __attribute__((ext_vector_type(4)));

// ---------------------------------------------------------------------------
// Kernel 1: numerator score per batch element (gather/reduce, memory-trivial)
// ---------------------------------------------------------------------------
__global__ __launch_bounds__(256) void crf_score_kernel(
    const float* __restrict__ em, const int* __restrict__ tags,
    const unsigned char* __restrict__ mask,
    const float* __restrict__ startt, const float* __restrict__ endt,
    const float* __restrict__ trans, float* __restrict__ score_out) {
  int b = blockIdx.x * blockDim.x + threadIdx.x;
  if (b >= BATCH) return;
  int t0 = tags[b];
  float sc = startt[t0] + em[(size_t)b * NT + t0];
  int prev = t0;
  int cnt = mask[b] ? 1 : 0;
  for (int l = 1; l < SEQL; ++l) {
    int t = tags[(size_t)l * BATCH + b];
    unsigned char m = mask[(size_t)l * BATCH + b];
    float v = trans[prev * NT + t] + em[(size_t)l * BATCH * NT + (size_t)b * NT + t];
    if (m) { sc += v; cnt++; }
    prev = t;
  }
  int last = tags[(size_t)(cnt - 1) * BATCH + b];
  sc += endt[last];
  score_out[b] = sc;
}

// ---------------------------------------------------------------------------
// Kernel 2: forward scan (partition function) via exp-domain WMMA matmul.
// One block = 16 batch rows; 3 waves, wave w owns output columns [16w,16w+16).
// Per step (3 barriers): p = exp(alpha - rowmax) in f16 ->
// (16x64) @ expT(64x48) with two INDEPENDENT v_wmma_f32_16x16x32_f16 (summed),
// then alpha = em + rowmax + ln2*log2(acc).
// ---------------------------------------------------------------------------
__global__ __launch_bounds__(96) void crf_scan_kernel(
    const float* __restrict__ em, const unsigned char* __restrict__ mask,
    const float* __restrict__ startt, const float* __restrict__ endt,
    const float* __restrict__ trans, float* __restrict__ denom_out) {
  __shared__ float    alpha_s[16][NT];                   // current alpha (f32)
  __shared__ _Float16 pbuf[16][64];                      // exp(alpha-m), K pad 64
  __shared__ __attribute__((aligned(16))) float mrow[16];// per-row max
  __shared__ float    pmax[16][6];                       // partial row maxima

  const int tid  = threadIdx.x;
  const int lane = tid & 31;
  const int w    = tid >> 5;       // wave id 0..2 -> N tile
  const int ln   = lane & 15;
  const int hi   = lane >> 4;
  const int hi8  = hi * 8;
  const int b0   = blockIdx.x * 16;
  const int j    = 16 * w + ln;    // global column (tag) for B and C/D

  // --- Build exp(T) as B operand (32x16 f16, contiguous k = 16*hi + c, same as A) once ---
  half16 B0, B1;
  #pragma unroll
  for (int c = 0; c < 16; ++c) {
    int i0 = (hi ? 16 : 0) + c;                 // K rows 0..31
    B0[c] = (_Float16)__expf(trans[i0 * NT + j]);
    int i1 = 32 + (hi ? 16 : 0) + c;            // K rows 32..63 (pad >=48)
    B1[c] = (i1 < NT) ? (_Float16)__expf(trans[i1 * NT + j]) : (_Float16)0.0f;
  }

  // --- Init alpha0 = start + emissions[0], and zero the K-pad of pbuf ---
  for (int e = tid; e < 16 * NT; e += 96) {
    int row = e / NT, col = e % NT;
    alpha_s[row][col] = startt[col] + em[(size_t)(b0 + row) * NT + col];
  }
  for (int e = tid; e < 16 * 16; e += 96)
    pbuf[e >> 4][48 + (e & 15)] = (_Float16)0.0f;
  __syncthreads();

  // Register copy of this lane's alpha tile (C/D layout: row = r + 8*hi, col j)
  float aregs[8];
  #pragma unroll
  for (int r = 0; r < 8; ++r) aregs[r] = alpha_s[r + hi8][j];

  // Prefetch emissions for step 1
  float em_cur[8];
  #pragma unroll
  for (int r = 0; r < 8; ++r)
    em_cur[r] = em[(size_t)1 * BATCH * NT + (size_t)(b0 + r + hi8) * NT + j];

  const int prow_r = ln;              // A-matrix row = lane&15
  const float LN2 = 0.69314718055994531f;
  const int row96 = tid & 15, seg96 = tid >> 4;   // 16 rows x 6 segments

  for (int l = 1; l < SEQL; ++l) {
    // Software prefetch for next step (overlaps with WMMA chain below)
    float em_nxt[8];
    int lp = (l < SEQL - 1) ? (l + 1) : 1;
    #pragma unroll
    for (int r = 0; r < 8; ++r)
      em_nxt[r] = em[(size_t)lp * BATCH * NT + (size_t)(b0 + r + hi8) * NT + j];
    // 8 mask bytes for rows hi8..hi8+7 as two aligned 4-byte loads
    unsigned mlo = *(const unsigned*)(mask + (size_t)l * BATCH + b0 + hi8);
    unsigned mhi = *(const unsigned*)(mask + (size_t)l * BATCH + b0 + hi8 + 4);

    // --- partial row max: 96 threads, 8 cols each ---
    __syncthreads();                                       // (1) alpha visible
    {
      float m = -3.0e38f;
      #pragma unroll
      for (int c2 = 0; c2 < 8; ++c2) m = fmaxf(m, alpha_s[row96][seg96 * 8 + c2]);
      pmax[row96][seg96] = m;
    }
    __syncthreads();                                       // (2) pmax visible
    // --- fused combine + p = exp(alpha - m) into f16 pbuf ---
    {
      float mr = pmax[row96][0];
      #pragma unroll
      for (int s = 1; s < 6; ++s) mr = fmaxf(mr, pmax[row96][s]);
      if (seg96 == 0) mrow[row96] = mr;                    // publish for phase D
      #pragma unroll
      for (int c2 = 0; c2 < 8; ++c2) {
        int c = seg96 * 8 + c2;
        pbuf[row96][c] = (_Float16)__expf(alpha_s[row96][c] - mr);
      }
    }
    __syncthreads();                                       // (3) pbuf/mrow visible

    // --- A-layout loads (16-byte LDS reads) + two independent WMMAs ---
    const half8* prow = (const half8*)(&pbuf[prow_r][0]);
    half8 a0lo = prow[0 + 2 * hi], a0hi = prow[1 + 2 * hi];
    half8 a1lo = prow[4 + 2 * hi], a1hi = prow[5 + 2 * hi];
    half16 A0 = __builtin_shufflevector(a0lo, a0hi, 0,1,2,3,4,5,6,7,8,9,10,11,12,13,14,15);
    half16 A1 = __builtin_shufflevector(a1lo, a1hi, 0,1,2,3,4,5,6,7,8,9,10,11,12,13,14,15);
    float8 z = {0.f,0.f,0.f,0.f,0.f,0.f,0.f,0.f};
    float8 acc0 = __builtin_amdgcn_wmma_f32_16x16x32_f16(false, A0, false, B0, (short)0, z, false, false);
    float8 acc1 = __builtin_amdgcn_wmma_f32_16x16x32_f16(false, A1, false, B1, (short)0, z, false, false);
    float8 acc = acc0 + acc1;

    // mrow for my 8 rows: two 16-byte LDS vector loads
    const float4v* mp = (const float4v*)(&mrow[hi8]);
    float4v m0 = mp[0], m1 = mp[1];
    float mrr[8] = {m0.x, m0.y, m0.z, m0.w, m1.x, m1.y, m1.z, m1.w};

    // --- alpha update: branch-free masked select + unconditional store ---
    #pragma unroll
    for (int r = 0; r < 8; ++r) {
      float base = em_cur[r] + mrr[r];
      float nv = fmaf(LN2, __log2f(acc[r]), base);   // acc in [~0.9, ~53]
      unsigned mw = (r < 4) ? mlo : mhi;
      bool mv = ((mw >> (8 * (r & 3))) & 0xffu) != 0;
      aregs[r] = mv ? nv : aregs[r];
      alpha_s[r + hi8][j] = aregs[r];
      em_cur[r] = em_nxt[r];
    }
  }

  __syncthreads();
  // --- denominator = logsumexp(alpha + end_transitions) per row ---
  if (tid < 16) {
    int row = tid;
    float m2 = -3.0e38f;
    for (int c = 0; c < NT; ++c) m2 = fmaxf(m2, alpha_s[row][c] + endt[c]);
    float s = 0.0f;
    for (int c = 0; c < NT; ++c) s += __expf(alpha_s[row][c] + endt[c] - m2);
    denom_out[b0 + row] = m2 + __logf(s);
  }
}

// ---------------------------------------------------------------------------
// Kernel 3: deterministic mean of (score - denom)
// ---------------------------------------------------------------------------
__global__ __launch_bounds__(256) void crf_reduce_kernel(
    const float* __restrict__ score, const float* __restrict__ denom,
    float* __restrict__ out) {
  __shared__ float sdata[256];
  int t = threadIdx.x;
  float s = 0.0f;
  for (int b = t; b < BATCH; b += 256) s += score[b] - denom[b];
  sdata[t] = s;
  __syncthreads();
  for (int k = 128; k > 0; k >>= 1) {
    if (t < k) sdata[t] += sdata[t + k];
    __syncthreads();
  }
  if (t == 0) out[0] = sdata[0] / (float)BATCH;
}

// ---------------------------------------------------------------------------
extern "C" void kernel_launch(void* const* d_in, const int* in_sizes, int n_in,
                              void* d_out, int out_size, void* d_ws, size_t ws_size,
                              hipStream_t stream) {
  const float*         em     = (const float*)d_in[0];
  const int*           tags   = (const int*)d_in[1];
  const unsigned char* mask   = (const unsigned char*)d_in[2];
  const float*         startt = (const float*)d_in[3];
  const float*         endt   = (const float*)d_in[4];
  const float*         trans  = (const float*)d_in[5];

  float* score = (float*)d_ws;           // [BATCH]
  float* denom = score + BATCH;          // [BATCH]

  crf_score_kernel<<<(BATCH + 255) / 256, 256, 0, stream>>>(
      em, tags, mask, startt, endt, trans, score);
  crf_scan_kernel<<<BATCH / 16, 96, 0, stream>>>(
      em, mask, startt, endt, trans, denom);
  crf_reduce_kernel<<<1, 256, 0, stream>>>(score, denom, (float*)d_out);
}
